// CrossAttention_20770461844123
// MI455X (gfx1250) — matrix operand order repaired and checked by reading, not compiled, not run on hardware
//
#include <hip/hip_runtime.h>
#include <hip/hip_bf16.h>

typedef __attribute__((ext_vector_type(16))) _Float16 v16h;
typedef __attribute__((ext_vector_type(8)))  _Float16 v8h;
typedef __attribute__((ext_vector_type(8)))  float    v8f;
typedef __attribute__((ext_vector_type(4)))  float    v4f;

union Frag { v16h v; v8h h[2]; };

#define NB        4
#define NN        16384
#define NC        256
#define NH        8
#define HD        32
#define SCALE     0.17677669529663687f

// ---------------- zero init for Gram accumulators ----------------
__global__ void zero_kernel(float* __restrict__ p, int n) {
    int i = blockIdx.x * blockDim.x + threadIdx.x;
    if (i < n) p[i] = 0.f;
}

// ---------------- Pass A: per-batch Gram  G = X^T X --------------
// grid: 256 = (stream*4+b)[8] x chunk[32]; block: 512 threads (16 waves)
#define CHUNK_ROWS 512
#define SUB_ROWS   64
#define NSUB       (CHUNK_ROWS / SUB_ROWS)
#define LDS_STRIDE 72   // halves per channel row (pad: conflict-avoiding)

__global__ __launch_bounds__(512)
void gram_kernel(const float* __restrict__ rgb, const float* __restrict__ xin,
                 float* __restrict__ G /* [2][4][256][256] */) {
    int blk    = blockIdx.x;
    int chunk  = blk & 31;
    int sb     = blk >> 5;            // stream*4 + b
    int stream = sb >> 2, b = sb & 3;
    const float* X    = (stream ? xin : rgb) + (size_t)b * NN * NC;
    float*       Gout = G + (size_t)sb * NC * NC;

    __shared__ _Float16 lds[2][NC * LDS_STRIDE];   // 2 x 36 KiB (double buffer)

    int t    = threadIdx.x;
    int lane = t & 31;
    int wave = t >> 5;          // 0..15  -> tile row
    int hi   = lane >> 4;       // 0/1
    int lcol = lane & 15;

    v8f acc[16];
    v8f zero = {0.f, 0.f, 0.f, 0.f, 0.f, 0.f, 0.f, 0.f};
#pragma unroll
    for (int i = 0; i < 16; ++i) acc[i] = zero;

    int nbase = chunk * CHUNK_ROWS;
    int c  = t & 255;           // channel handled by this thread in staging
    int p0 = t >> 8;            // 0..1 row-pair phase

    float ra[16], rb[16];       // in-flight f32 rows for next subchunk

    auto load_sub = [&](int sub_i) {
        const float* src = X + (size_t)(nbase + sub_i * SUB_ROWS) * NC + c;
#pragma unroll
        for (int i = 0; i < 16; ++i) {
            int p = p0 + 2 * i;                      // row-pair 0..31
            ra[i] = src[(size_t)(2 * p + 0) * NC];
            rb[i] = src[(size_t)(2 * p + 1) * NC];
        }
    };
    auto store_sub = [&](int bufi) {
#pragma unroll
        for (int i = 0; i < 16; ++i) {
            int p = p0 + 2 * i;
            union { _Float16 h[2]; unsigned int u; } pk;
            pk.h[0] = (_Float16)ra[i];
            pk.h[1] = (_Float16)rb[i];
            *(unsigned int*)(&lds[bufi][c * LDS_STRIDE + 2 * p]) = pk.u;
        }
    };

    // prologue: stage subchunk 0 into buffer 0
    load_sub(0);
    store_sub(0);
    __syncthreads();

    for (int sub = 0; sub < NSUB; ++sub) {
        int cur = sub & 1;
        // kick off global loads for the next subchunk before computing
        if (sub + 1 < NSUB) load_sub(sub + 1);

        const _Float16* buf = &lds[cur][0];
        int ci = wave * 16;      // A-channel base (tile row)
#pragma unroll
        for (int ks = 0; ks < 2; ++ks) {
            int nb = ks * 32;
            // A-frag (16x32 f16): lane=M, 16 consecutive K from hi*16
            const _Float16* ap = &buf[(ci + lcol) * LDS_STRIDE + nb + hi * 16];
            Frag A;
            A.h[0] = *(const v8h*)(ap);
            A.h[1] = *(const v8h*)(ap + 8);
            // B-frag (32x16 f16): lane=N, 16 consecutive K from hi*16
            const _Float16* bbase = &buf[lcol * LDS_STRIDE + nb + hi * 16];
            Frag Bc;
            Bc.h[0] = *(const v8h*)(bbase);
            Bc.h[1] = *(const v8h*)(bbase + 8);
#pragma unroll
            for (int jt = 0; jt < 16; ++jt) {
                Frag Bn = Bc;
                if (jt < 15) {   // software pipeline: next B before current WMMA
                    const _Float16* bp = bbase + (size_t)(jt + 1) * 16 * LDS_STRIDE;
                    Bn.h[0] = *(const v8h*)(bp);
                    Bn.h[1] = *(const v8h*)(bp + 8);
                }
                acc[jt] = __builtin_amdgcn_wmma_f32_16x16x32_f16(
                    false, A.v, false, Bc.v, (short)0, acc[jt], false, false);
                Bc = Bn;
            }
        }

        // convert + store next subchunk into the other buffer
        if (sub + 1 < NSUB) store_sub(1 - cur);
        __syncthreads();
    }

    // writeback: D layout VGPR r -> M=r+8*hi, lane -> N=lcol
#pragma unroll
    for (int jt = 0; jt < 16; ++jt) {
        int col     = jt * 16 + lcol;
        int rowbase = wave * 16 + hi * 8;
#pragma unroll
        for (int r = 0; r < 8; ++r) {
            atomicAdd(&Gout[(size_t)(rowbase + r) * NC + col], acc[jt][r]);
        }
    }
}

// ------- Pass B: ctx = softmax_d( scale * Wk G Wv^T ), tiny ------
// grid: 64 = (stream*4+b)*8 + h ; block 256
__global__ __launch_bounds__(256)
void ctx_kernel(const float* __restrict__ G,
                const float* __restrict__ Wkv_rgb, const float* __restrict__ Wkv_x,
                _Float16* __restrict__ ctxT /* [2][4][8][32 e][32 d] f16 */) {
    int blk = blockIdx.x;
    int h   = blk & 7;
    int sb  = blk >> 3;
    int stream = sb >> 2;
    const float* Gm = G + (size_t)sb * NC * NC;
    const float* W  = stream ? Wkv_x : Wkv_rgb;
    const float* Wk = W + (size_t)(h * HD) * NC;
    const float* Wv = W + (size_t)(NC + h * HD) * NC;

    __shared__ float T[HD][NC];    // 32 KiB
    __shared__ float pre[HD][HD];  // 4 KiB

    int t = threadIdx.x;
    // step 1: T[d][c2] = sum_c1 Wk[d][c1] * G[c1][c2]
    {
        int d   = t >> 3;
        int c2b = (t & 7) * 32;
        float accv[32];
#pragma unroll
        for (int j = 0; j < 32; ++j) accv[j] = 0.f;
        for (int c1 = 0; c1 < NC; ++c1) {
            float w = Wk[(size_t)d * NC + c1];
            const float* grow = Gm + (size_t)c1 * NC + c2b;
#pragma unroll
            for (int j = 0; j < 32; ++j) accv[j] += w * grow[j];
        }
#pragma unroll
        for (int j = 0; j < 32; ++j) T[d][c2b + j] = accv[j];
    }
    __syncthreads();
    // step 2: pre[d][e] = scale * sum_c2 T[d][c2] * Wv[e][c2]
    {
        int d  = t >> 3;
        int e0 = (t & 7) * 4;
        for (int e = e0; e < e0 + 4; ++e) {
            float acc = 0.f;
            const float* wr = Wv + (size_t)e * NC;
            for (int c2 = 0; c2 < NC; ++c2) acc += T[d][c2] * wr[c2];
            pre[d][e] = acc * SCALE;
        }
    }
    __syncthreads();
    // step 3: column softmax over d; store transposed [e][d] as f16
    if (t < HD) {
        int e = t;
        float m = -1e30f;
        for (int d = 0; d < HD; ++d) m = fmaxf(m, pre[d][e]);
        float ex[HD], s = 0.f;
        for (int d = 0; d < HD; ++d) { ex[d] = __expf(pre[d][e] - m); s += ex[d]; }
        float inv = 1.f / s;
        _Float16* dst = ctxT + (size_t)(sb * NH + h) * (HD * HD) + e * HD;
        for (int d = 0; d < HD; ++d) dst[d] = (_Float16)(ex[d] * inv);
    }
}

// ---------------- Pass C: out = Q @ blockdiag(ctx_cross) ----------
// grid: 1024 = (stream*4+b)[8] x chunk[128]; block 256 (8 waves, 16 rows each)
__global__ __launch_bounds__(256)
void out_kernel(const float* __restrict__ rgb, const float* __restrict__ xin,
                const _Float16* __restrict__ ctxT, float* __restrict__ out) {
    int blk    = blockIdx.x;
    int chunk  = blk & 127;
    int sb     = blk >> 7;
    int stream = sb >> 2, b = sb & 3;
    const float* Q = (stream ? xin : rgb) + (size_t)b * NN * NC;
    // cross wiring: out_rgb uses ctx_x, out_x uses ctx_rgb
    const _Float16* Cx = ctxT + (size_t)(((1 - stream) * 4 + b) * NH) * (HD * HD);
    float* O = out + (size_t)stream * NB * NN * NC + (size_t)b * NN * NC;

    int t = threadIdx.x, lane = t & 31, wave = t >> 5;
    int hi = lane >> 4, lcol = lane & 15;
    int nrow = chunk * 128 + wave * 16;

    v8f zero = {0.f, 0.f, 0.f, 0.f, 0.f, 0.f, 0.f, 0.f};

    // A-frag source: Q[nrow+lcol, h*32 + d], runs d = hi*8..+7 and 16+hi*8..+7
    const float* qbase = Q + (size_t)(nrow + lcol) * NC + hi * 8;
    v4f q0 = *(const v4f*)(qbase);
    v4f q1 = *(const v4f*)(qbase + 4);
    v4f q2 = *(const v4f*)(qbase + 16);
    v4f q3 = *(const v4f*)(qbase + 20);

#pragma unroll
    for (int h = 0; h < NH; ++h) {
        v4f n0 = q0, n1 = q1, n2 = q2, n3 = q3;
        if (h + 1 < NH) {  // software pipeline: prefetch next head's Q
            const float* np = qbase + (size_t)(h + 1) * HD;
            n0 = *(const v4f*)(np);
            n1 = *(const v4f*)(np + 4);
            n2 = *(const v4f*)(np + 16);
            n3 = *(const v4f*)(np + 20);
        }
        Frag A;
#pragma unroll
        for (int k = 0; k < 4; ++k) {
            A.v[k]      = (_Float16)q0[k];
            A.v[4 + k]  = (_Float16)q1[k];
            A.v[8 + k]  = (_Float16)q2[k];
            A.v[12 + k] = (_Float16)q3[k];
        }
        const _Float16* ch = Cx + (size_t)h * (HD * HD);
        // load both e-tiles' B-frags up front
        Frag B0, B1;
        {
            const _Float16* bp0 = ch + (lcol) * HD + hi * 8;
            const _Float16* bp1 = ch + (16 + lcol) * HD + hi * 8;
            B0.h[0] = *(const v8h*)(bp0);
            B0.h[1] = *(const v8h*)(bp0 + 16);
            B1.h[0] = *(const v8h*)(bp1);
            B1.h[1] = *(const v8h*)(bp1 + 16);
        }
        v8f o0 = __builtin_amdgcn_wmma_f32_16x16x32_f16(
            false, A.v, false, B0.v, (short)0, zero, false, false);
        v8f o1 = __builtin_amdgcn_wmma_f32_16x16x32_f16(
            false, A.v, false, B1.v, (short)0, zero, false, false);
        float* op = O + (size_t)(nrow + hi * 8) * NC + h * HD + lcol;
#pragma unroll
        for (int r = 0; r < 8; ++r) op[(size_t)r * NC]      = o0[r];
#pragma unroll
        for (int r = 0; r < 8; ++r) op[(size_t)r * NC + 16] = o1[r];
        q0 = n0; q1 = n1; q2 = n2; q3 = n3;
    }
}

extern "C" void kernel_launch(void* const* d_in, const int* in_sizes, int n_in,
                              void* d_out, int out_size, void* d_ws, size_t ws_size,
                              hipStream_t stream) {
    const float* rgb = (const float*)d_in[0];
    const float* x   = (const float*)d_in[1];
    const float* Wr  = (const float*)d_in[2];
    const float* Wx  = (const float*)d_in[3];
    float* out = (float*)d_out;

    float*    G    = (float*)d_ws;                                      // 2 MiB
    _Float16* ctxT = (_Float16*)((char*)d_ws + (size_t)8 * NC * NC * 4); // 128 KiB

    const int gN = 8 * NC * NC;
    zero_kernel<<<(gN + 255) / 256, 256, 0, stream>>>(G, gN);
    gram_kernel<<<256, 512, 0, stream>>>(rgb, x, G);
    ctx_kernel<<<64, 256, 0, stream>>>(G, Wr, Wx, ctxT);
    out_kernel<<<1024, 256, 0, stream>>>(rgb, x, ctxT, out);
}
